// GRUDecoder_69801808494928
// MI455X (gfx1250) — matrix operand order repaired and checked by reading, not compiled, not run on hardware
//
#include <hip/hip_runtime.h>

// GRU decoder for MI455X (gfx1250): B=2048, F=128, H=512, T=96.
// Compute-bound (~413 GFLOP; HBM side is only ~100MB of output writes =
// ~4.3us at 23.3 TB/s) -> all three matmuls per step run on
// v_wmma_f32_16x16x32_bf16. Batch rows recur independently, so each
// workgroup owns a 32-row batch tile and iterates all 96 steps locally.
// M=32 per WG (two 16-row subtiles sharing every B weight fragment in
// registers) keeps the L2 weight stream at 64 WGs x 2MB bf16/step
// (~12.3 GB total, 32 FLOP per B-byte). Gate nonlinearities use raw
// v_exp_f32 / v_rcp_f32 to keep the barrier-bracketed epilogue short.

#define BATCH  2048
#define FDIM   128
#define HDIM   512
#define TSTEPS 96
#define G3H    1536   // 3*H

typedef __attribute__((ext_vector_type(16))) __bf16 v16bf;
typedef __attribute__((ext_vector_type(8)))  __bf16 v8bf;
typedef __attribute__((ext_vector_type(8)))  float  v8f;

__device__ __forceinline__ unsigned short f32_to_bf16(float f) {
  unsigned int u = __float_as_uint(f);
  u += 0x7FFFu + ((u >> 16) & 1u);     // round-to-nearest-even
  return (unsigned short)(u >> 16);
}

// sigmoid via one v_exp_f32 + one raw v_rcp_f32
__device__ __forceinline__ float sigm(float x) {
  return __builtin_amdgcn_rcpf(1.0f + __expf(-x));
}

// tanh(x) = 2*sigmoid(2x) - 1  (one v_exp + one v_rcp + fma)
__device__ __forceinline__ float fast_tanh(float x) {
  return __builtin_fmaf(2.0f, sigm(2.0f * x), -1.0f);
}

__device__ __forceinline__ v16bf cat16(v8bf lo, v8bf hi) {
  v16bf r;
#pragma unroll
  for (int i = 0; i < 8; ++i) { r[i] = lo[i]; r[i + 8] = hi[i]; }
  return r;
}

// A fragment (16xK tile, row-major bf16 in LDS).
// 16-bit A layout: lanes 0-15 hold M=lane, K = k0+[0..7] then k0+16+[0..7];
// lanes 16-31 hold M=lane-16, K = k0+8+[0..7] then k0+24+[0..7].
__device__ __forceinline__ v16bf load_A(const unsigned short* lds, int stride,
                                        int k0, int lane) {
  const unsigned short* p = lds + (lane & 15) * stride + k0 + ((lane >> 4) << 3);
  v8bf lo = *(const v8bf*)(p);        // two ds_load_b128
  v8bf hi = *(const v8bf*)(p + 16);
  return cat16(lo, hi);
}

// B fragment of w^T[k0..k0+32, col..col+16] where w is row-major [Nout][Ktot].
// 16-bit B layout: lane holds N = lane%16, K = (lane/16)*8 + [0..7] then
// 16 + (lane/16)*8 + [0..7] in w's row -> two global_load_b128, L2-resident weights.
__device__ __forceinline__ v16bf load_B(const unsigned short* __restrict__ w,
                                        int Ktot, int col, int k0, int lane) {
  const unsigned short* p =
      w + (size_t)(col + (lane & 15)) * Ktot + k0 + ((lane >> 4) << 3);
  v8bf lo = *(const v8bf*)(p);
  v8bf hi = *(const v8bf*)(p + 16);
  return cat16(lo, hi);
}

#define WMMA(acc, a, b)                                                        \
  (acc) = __builtin_amdgcn_wmma_f32_16x16x32_bf16(                             \
      false, (a), false, (b), (short)0, (acc), false, false)

__global__ void cvt_bf16_kernel(const float* __restrict__ src,
                                unsigned short* __restrict__ dst, int n) {
  int i = blockIdx.x * blockDim.x + threadIdx.x;
  if (i < n) dst[i] = f32_to_bf16(src[i]);
}

__global__ __launch_bounds__(256) void gru_decoder_kernel(
    const float* __restrict__ hidden,           // [B][H] f32
    const unsigned short* __restrict__ wih,     // [3H][F] bf16
    const unsigned short* __restrict__ whh,     // [3H][H] bf16
    const float* __restrict__ b_ih,             // [3H]
    const float* __restrict__ b_hh,             // [3H]
    const unsigned short* __restrict__ fcw,     // [F][H] bf16
    const float* __restrict__ fc_b,             // [F]
    float* __restrict__ out) {                  // [B][T][F] f32 (time-reversed)
  constexpr int MROWS = 32;  // batch rows per WG (2 WMMA subtiles)
  constexpr int HBS   = 520; // padded bf16 row stride for the h tile
  constexpr int HFS   = 516; // padded f32 row stride for the h master copy

  __shared__ unsigned short sh_hb[2][MROWS * HBS];  // double-buffered bf16 h
  __shared__ float          sh_hf[MROWS * HFS];     // f32 h master copy
  __shared__ unsigned short sh_x[MROWS * FDIM];     // bf16 x tile
  __shared__ float sh_bih[G3H];
  __shared__ float sh_bhh[G3H];
  __shared__ float sh_fcb[FDIM];

  const int tid   = threadIdx.x;
  const int lane  = tid & 31;
  const int wave  = tid >> 5;       // 8 waves
  const int bbase = blockIdx.x * MROWS;

  // Preload biases, zero x0, stage h0 (f32 master + bf16 operand buffer 0).
  for (int i = tid; i < G3H; i += 256) { sh_bih[i] = b_ih[i]; sh_bhh[i] = b_hh[i]; }
  if (tid < FDIM) sh_fcb[tid] = fc_b[tid];
  for (int i = tid; i < MROWS * FDIM; i += 256) sh_x[i] = 0;
  for (int i = tid; i < MROWS * HDIM; i += 256) {
    int r = i >> 9, c = i & (HDIM - 1);
    float hv = hidden[(size_t)(bbase + r) * HDIM + c];
    sh_hf[r * HFS + c] = hv;
    sh_hb[0][r * HBS + c] = f32_to_bf16(hv);
  }
  __syncthreads();

  for (int t = 0; t < TSTEPS; ++t) {
    const int cur = t & 1;            // read h_t from cur, write h_{t+1} to nxt
    const unsigned short* hb_rd = sh_hb[cur];
    unsigned short* hb_wr = sh_hb[cur ^ 1];

    // ---- gate matmuls: each wave owns 4 hidden-column tiles of 16;
    //      each B fragment is reused for both 16-row M-subtiles ----
#pragma unroll
    for (int jj = 0; jj < 4; ++jj) {
      const int colh = (jj * 8 + wave) * 16;
      v8f ar[2] = {{}, {}}, az[2] = {{}, {}};   // r/z fold gi+gh together
      v8f ain[2] = {{}, {}}, ahn[2] = {{}, {}};

      // input-side contribution: K = F = 128 over x_{t-1}
#pragma unroll 2
      for (int k0 = 0; k0 < FDIM; k0 += 32) {
        v16bf a0 = load_A(sh_x, FDIM, k0, lane);
        v16bf a1 = load_A(sh_x + 16 * FDIM, FDIM, k0, lane);
        v16bf br = load_B(wih, FDIM, colh, k0, lane);
        WMMA(ar[0], a0, br);
        WMMA(ar[1], a1, br);
        v16bf bz = load_B(wih, FDIM, HDIM + colh, k0, lane);
        WMMA(az[0], a0, bz);
        WMMA(az[1], a1, bz);
        v16bf bn = load_B(wih, FDIM, 2 * HDIM + colh, k0, lane);
        WMMA(ain[0], a0, bn);
        WMMA(ain[1], a1, bn);
      }
      // hidden-side contribution: K = H = 512 over h_{t-1}
#pragma unroll 2
      for (int k0 = 0; k0 < HDIM; k0 += 32) {
        v16bf a0 = load_A(hb_rd, HBS, k0, lane);
        v16bf a1 = load_A(hb_rd + 16 * HBS, HBS, k0, lane);
        v16bf br = load_B(whh, HDIM, colh, k0, lane);
        WMMA(ar[0], a0, br);
        WMMA(ar[1], a1, br);
        v16bf bz = load_B(whh, HDIM, HDIM + colh, k0, lane);
        WMMA(az[0], a0, bz);
        WMMA(az[1], a1, bz);
        v16bf bn = load_B(whh, HDIM, 2 * HDIM + colh, k0, lane);
        WMMA(ahn[0], a0, bn);
        WMMA(ahn[1], a1, bn);
      }

      // gate epilogue entirely in the C-fragment layout
      const int col   = colh + (lane & 15);
      const float bir = sh_bih[col],            bhr = sh_bhh[col];
      const float biz = sh_bih[HDIM + col],     bhz = sh_bhh[HDIM + col];
      const float bin = sh_bih[2 * HDIM + col], bhn = sh_bhh[2 * HDIM + col];
#pragma unroll
      for (int s = 0; s < 2; ++s) {
#pragma unroll
        for (int i = 0; i < 8; ++i) {
          const int row = s * 16 + i + ((lane >> 4) << 3);
          const float rg = sigm(ar[s][i] + bir + bhr);
          const float zg = sigm(az[s][i] + biz + bhz);
          const float ng = fast_tanh(ain[s][i] + bin + rg * (ahn[s][i] + bhn));
          const float ho = sh_hf[row * HFS + col];   // exclusive per-wave cell
          const float hn = (1.0f - zg) * ng + zg * ho;
          sh_hf[row * HFS + col] = hn;
          hb_wr[row * HBS + col] = f32_to_bf16(hn);
        }
      }
    }
    __syncthreads();  // h_t complete in the nxt buffer

    // ---- fc: x_t = h_t @ fc_w^T + fc_b ; one 16-col tile per wave,
    //      B fragment again shared by both M-subtiles ----
    {
      const int colf0 = wave * 16;
      v8f ax[2] = {{}, {}};
#pragma unroll 2
      for (int k0 = 0; k0 < HDIM; k0 += 32) {
        v16bf a0 = load_A(hb_wr, HBS, k0, lane);
        v16bf a1 = load_A(hb_wr + 16 * HBS, HBS, k0, lane);
        v16bf b  = load_B(fcw, HDIM, colf0, k0, lane);
        WMMA(ax[0], a0, b);
        WMMA(ax[1], a1, b);
      }
      const int colf = colf0 + (lane & 15);
      const float fb = sh_fcb[colf];
      const size_t tq = (size_t)(TSTEPS - 1 - t) * FDIM;  // reversed time
#pragma unroll
      for (int s = 0; s < 2; ++s) {
#pragma unroll
        for (int i = 0; i < 8; ++i) {
          const int row = s * 16 + i + ((lane >> 4) << 3);
          const float xv = ax[s][i] + fb;
          out[(size_t)(bbase + row) * (TSTEPS * FDIM) + tq + colf] = xv;
          sh_x[row * FDIM + colf] = f32_to_bf16(xv);
        }
      }
    }
    __syncthreads();  // x_t visible to next step's gate phase
  }
}

extern "C" void kernel_launch(void* const* d_in, const int* in_sizes, int n_in,
                              void* d_out, int out_size, void* d_ws, size_t ws_size,
                              hipStream_t stream) {
  const float* hidden = (const float*)d_in[0];
  const float* w_ih   = (const float*)d_in[1];  // [3H][F]
  const float* w_hh   = (const float*)d_in[2];  // [3H][H]
  const float* b_ih   = (const float*)d_in[3];
  const float* b_hh   = (const float*)d_in[4];
  const float* fc_w   = (const float*)d_in[5];  // [F][H]
  const float* fc_b   = (const float*)d_in[6];
  float* out = (float*)d_out;

  // bf16 weight cache in workspace (~2 MB, L2-resident afterwards)
  unsigned short* ws    = (unsigned short*)d_ws;
  unsigned short* wih_b = ws;
  unsigned short* whh_b = wih_b + G3H * FDIM;
  unsigned short* fcw_b = whh_b + G3H * HDIM;

  const int n1 = G3H * FDIM, n2 = G3H * HDIM, n3 = FDIM * HDIM;
  cvt_bf16_kernel<<<(n1 + 255) / 256, 256, 0, stream>>>(w_ih, wih_b, n1);
  cvt_bf16_kernel<<<(n2 + 255) / 256, 256, 0, stream>>>(w_hh, whh_b, n2);
  cvt_bf16_kernel<<<(n3 + 255) / 256, 256, 0, stream>>>(fc_w, fcw_b, n3);

  gru_decoder_kernel<<<BATCH / 32, 256, 0, stream>>>(
      hidden, wih_b, whh_b, b_ih, b_hh, fcw_b, fc_b, out);
}
